// CausalTransformerEncoder_8409545966071
// MI455X (gfx1250) — compile-verified, matrix operand order checked by reading, not run on hardware
//
#include <hip/hip_runtime.h>
#include <hip/hip_bf16.h>
#include <float.h>

// Problem constants (match reference)
#define B_  2
#define T_  2048
#define D_  1024
#define H_  16
#define L_  4
#define FF_ 4096
#define DH_ 64
#define M_  (B_*T_)     // 4096 rows in the flattened activation matrix
#define D3_ (3*D_)      // 3072

typedef unsigned short u16;
typedef unsigned int   u32;
typedef __attribute__((ext_vector_type(16))) __bf16 v16bf;
typedef __attribute__((ext_vector_type(8)))  float  v8f;
typedef __attribute__((ext_vector_type(8)))  u32    v8u;

// ---------------- helpers ----------------

__device__ __forceinline__ u16 f2bf(float f) {
  u32 u = __builtin_bit_cast(u32, f);
  u32 r = u + 0x7FFFu + ((u >> 16) & 1u);   // round-to-nearest-even
  return (u16)(r >> 16);
}

__device__ __forceinline__ v16bf make_frag(uint4 lo, uint4 hi) {
  v8u v = { lo.x, lo.y, lo.z, lo.w, hi.x, hi.y, hi.z, hi.w };
  return __builtin_bit_cast(v16bf, v);
}

// Gather one 16x32 bf16 matrix fragment (ISA 16-bit A-matrix layout):
// lane&15 -> row; lanes 0-15 hold K = [0..7] and [16..23], lanes 16-31 hold
// K = [8..15] and [24..31]; each half is one b128 load.
__device__ __forceinline__ v16bf load_frag(const u16* p, int pitch) {
  int lane = threadIdx.x & 31;
  const u16* q = p + (lane & 15) * pitch + ((lane >> 4) << 3);
  uint4 lo = *(const uint4*)q;
  uint4 hi = *(const uint4*)(q + 16);
  return make_frag(lo, hi);
}

__device__ __forceinline__ v8f wmma_bf16(v16bf a, v16bf b, v8f c) {
  // (neg_a, A, neg_b, B, c_mod, C, reuse_a, reuse_b)
  return __builtin_amdgcn_wmma_f32_16x16x32_bf16(false, a, false, b, (short)0, c,
                                                 false, false);
}

// Async global -> LDS copy of 16 bytes per lane (CDNA5 GLOBAL_LOAD_ASYNC_TO_LDS,
// tracked by ASYNCcnt; no VGPR round trip). LDS byte offset = low 32 bits of
// the generic pointer (ISA: LDS aperture address is addr[31:0]).
__device__ __forceinline__ void async_copy_b128(const u16* gsrc, u16* lds_dst) {
  u32 ldsoff = (u32)(unsigned long long)lds_dst;
  asm volatile("global_load_async_to_lds_b128 %0, %1, off"
               :: "v"(ldsoff), "v"(gsrc) : "memory");
}
__device__ __forceinline__ void wait_async_0() {
  asm volatile("s_wait_asynccnt 0x0" ::: "memory");
}
// Async loads complete in order: <=4 outstanding means the previous group of 4
// (the current tile's copies) has fully landed in LDS.
__device__ __forceinline__ void wait_async_4() {
  asm volatile("s_wait_asynccnt 0x4" ::: "memory");
}

// ---------------- LayerNorm kernels ----------------

// ln0 + sinusoidal positional encoding, f32 out (residual stream init)
__global__ __launch_bounds__(256)
void ln0_pe_kernel(const float* __restrict__ xin, const float* __restrict__ g,
                   const float* __restrict__ bta, float* __restrict__ xout) {
  __shared__ float sb[256];
  int row = blockIdx.x, tid = threadIdx.x;
  const float* xr = xin + (size_t)row * D_;
  float v[4];
  float s = 0.f;
#pragma unroll
  for (int j = 0; j < 4; ++j) { v[j] = xr[tid + 256 * j]; s += v[j]; }
  sb[tid] = s; __syncthreads();
  for (int st = 128; st > 0; st >>= 1) { if (tid < st) sb[tid] += sb[tid + st]; __syncthreads(); }
  float mu = sb[0] * (1.f / D_); __syncthreads();
  float s2 = 0.f;
#pragma unroll
  for (int j = 0; j < 4; ++j) { float d = v[j] - mu; s2 += d * d; }
  sb[tid] = s2; __syncthreads();
  for (int st = 128; st > 0; st >>= 1) { if (tid < st) sb[tid] += sb[tid + st]; __syncthreads(); }
  float rstd = rsqrtf(sb[0] * (1.f / D_) + 1e-5f);
  int t = row % T_;
#pragma unroll
  for (int j = 0; j < 4; ++j) {
    int c = tid + 256 * j;
    float y = (v[j] - mu) * rstd * g[c] + bta[c];
    int i2 = c & ~1;
    float freq = __expf(-(float)i2 * (9.2103403719761836f / (float)D_));
    float ang = (float)t * freq;
    y += (c & 1) ? __cosf(ang) : __sinf(ang);
    xout[(size_t)row * D_ + c] = y;
  }
}

// LayerNorm -> bf16 out (GEMM A operand)
__global__ __launch_bounds__(256)
void ln_bf_kernel(const float* __restrict__ x, const float* __restrict__ g,
                  const float* __restrict__ bta, u16* __restrict__ hout) {
  __shared__ float sb[256];
  int row = blockIdx.x, tid = threadIdx.x;
  const float* xr = x + (size_t)row * D_;
  float v[4];
  float s = 0.f;
#pragma unroll
  for (int j = 0; j < 4; ++j) { v[j] = xr[tid + 256 * j]; s += v[j]; }
  sb[tid] = s; __syncthreads();
  for (int st = 128; st > 0; st >>= 1) { if (tid < st) sb[tid] += sb[tid + st]; __syncthreads(); }
  float mu = sb[0] * (1.f / D_); __syncthreads();
  float s2 = 0.f;
#pragma unroll
  for (int j = 0; j < 4; ++j) { float d = v[j] - mu; s2 += d * d; }
  sb[tid] = s2; __syncthreads();
  for (int st = 128; st > 0; st >>= 1) { if (tid < st) sb[tid] += sb[tid + st]; __syncthreads(); }
  float rstd = rsqrtf(sb[0] * (1.f / D_) + 1e-5f);
#pragma unroll
  for (int j = 0; j < 4; ++j) {
    int c = tid + 256 * j;
    float y = (v[j] - mu) * rstd * g[c] + bta[c];
    hout[(size_t)row * D_ + c] = f2bf(y);
  }
}

// ---------------- weight transpose + f32->bf16 (once per GEMM) -------------
// W[k][n] (f32, KxN) -> Wt[n][k] (bf16, NxK row-major). Block: 32k x 64n.

__global__ __launch_bounds__(256)
void wt_kernel(const float* __restrict__ W, u16* __restrict__ Wt, int K, int N) {
  __shared__ u16 Lt[64][40];
  int k0 = blockIdx.y * 32, n0 = blockIdx.x * 64;
  int tid = threadIdx.x;
  int kr = tid >> 3, nc = (tid & 7) * 8;
  const float* wp = W + (size_t)(k0 + kr) * N + n0 + nc;
  float4 f0 = *(const float4*)wp;
  float4 f1 = *(const float4*)(wp + 4);
  u16 h[8] = { f2bf(f0.x), f2bf(f0.y), f2bf(f0.z), f2bf(f0.w),
               f2bf(f1.x), f2bf(f1.y), f2bf(f1.z), f2bf(f1.w) };
#pragma unroll
  for (int j = 0; j < 8; ++j) Lt[nc + j][kr] = h[j];
  __syncthreads();
  int flat = tid * 8;
  int n = flat >> 5, k = flat & 31;
  *(uint4*)(Wt + (size_t)(n0 + n) * K + k0 + k) = *(const uint4*)&Lt[n][k];
}

// ---------------- WMMA GEMM: C = A(bf16)[M,K] @ Wt(bf16)[N,K]^T + bias ------
// Double-buffered: async global->LDS copies for tile k+1 overlap the WMMA
// work on tile k (ASYNCcnt in-order completion makes the handoff a single
// s_wait_asynccnt 0x4).  GELU: exact gelu.  RES: +f32 residual, f32 out.

template <bool GELU, bool RES>
__global__ __launch_bounds__(256)
void gemm_bf16_kernel(const u16* __restrict__ A, const u16* __restrict__ Wt,
                      const float* __restrict__ bias, const float* __restrict__ resid,
                      float* __restrict__ outf, u16* __restrict__ outb,
                      int N, int K) {
  __shared__ u16 As[2][128][40];   // ping-pong [m][k] tiles (+8 halves pad)
  __shared__ u16 Bs[2][128][40];   // ping-pong [n][k] tiles
  int tid = threadIdx.x;
  int wave = tid >> 5, lane = tid & 31;
  int wm = wave >> 2, wn = wave & 3;           // 2x4 wave grid
  int blockN = blockIdx.x * 128, blockM = blockIdx.y * 128;
  v8f acc[4][2] = {};

  // per-thread tile-copy coordinates (two b128 chunks per tile per matrix)
  int r0 = (tid * 8) >> 5,         c0 = (tid * 8) & 31;
  int r1 = ((tid + 256) * 8) >> 5, c1 = ((tid + 256) * 8) & 31;
  const u16* a0 = A + (size_t)(blockM + r0) * K + c0;
  const u16* a1 = A + (size_t)(blockM + r1) * K + c1;
  const u16* b0 = Wt + (size_t)(blockN + r0) * K + c0;
  const u16* b1 = Wt + (size_t)(blockN + r1) * K + c1;

  int nk = K >> 5;
  // prologue: start tile 0 into buffer 0
  async_copy_b128(a0, &As[0][r0][c0]);
  async_copy_b128(a1, &As[0][r1][c1]);
  async_copy_b128(b0, &Bs[0][r0][c0]);
  async_copy_b128(b1, &Bs[0][r1][c1]);

  for (int kt = 0; kt < nk; ++kt) {
    int cur = kt & 1;
    if (kt + 1 < nk) {
      // other buffer was drained at end of iteration kt-1 (barrier below)
      int kn = (kt + 1) << 5;
      int nxt = cur ^ 1;
      async_copy_b128(a0 + kn, &As[nxt][r0][c0]);
      async_copy_b128(a1 + kn, &As[nxt][r1][c1]);
      async_copy_b128(b0 + kn, &Bs[nxt][r0][c0]);
      async_copy_b128(b1 + kn, &Bs[nxt][r1][c1]);
      if (kt + 2 < nk) {                   // keep L2 stream 2 tiles ahead
        __builtin_prefetch(a0 + ((kt + 2) << 5), 0, 1);
        __builtin_prefetch(b0 + ((kt + 2) << 5), 0, 1);
      }
      wait_async_4();                      // tile kt's 4 copies have landed
    } else {
      wait_async_0();
    }
    __syncthreads();                       // every wave's tile-kt copies done
    v16bf bf[2];
#pragma unroll
    for (int n = 0; n < 2; ++n) bf[n] = load_frag(&Bs[cur][wn * 32 + n * 16][0], 40);
#pragma unroll
    for (int m = 0; m < 4; ++m) {
      v16bf af = load_frag(&As[cur][wm * 64 + m * 16][0], 40);
      acc[m][0] = wmma_bf16(af, bf[0], acc[m][0]);
      acc[m][1] = wmma_bf16(af, bf[1], acc[m][1]);
    }
    __syncthreads();                       // buffer cur drained; reusable
  }

  // epilogue: C layout: VGPR r -> row r (+8 for lanes 16-31), lane&15 -> col
  int hi8 = (lane >> 4) << 3;
  int nlo = lane & 15;
#pragma unroll
  for (int n = 0; n < 2; ++n) {
    int col = blockN + wn * 32 + n * 16 + nlo;
    float bv = bias[col];
#pragma unroll
    for (int m = 0; m < 4; ++m) {
#pragma unroll
      for (int r = 0; r < 8; ++r) {
        int row = blockM + wm * 64 + m * 16 + r + hi8;
        float y = acc[m][n][r] + bv;
        if (GELU) y = 0.5f * y * (1.f + erff(y * 0.70710678118654752f));
        if (RES)  outf[(size_t)row * N + col] = y + resid[(size_t)row * N + col];
        else      outb[(size_t)row * N + col] = f2bf(y);
      }
    }
  }
}

// ---------------- Flash attention (causal + length mask) ----------------
// Block: (b, h, 64 query rows); 4 waves, 16 query rows each.
// qkv layout: [B*T, 3*D], q at col h*64+d, k at D+..., v at 2D+...

__global__ __launch_bounds__(128)
void attn_kernel(const u16* __restrict__ qkv, const int* __restrict__ lengths,
                 u16* __restrict__ obf) {
  __shared__ u16 Vt[64][40];        // [d][key] for current key tile
  __shared__ u16 Psc[4][16][40];    // per-wave P relayout scratch
  int tid = threadIdx.x, wave = tid >> 5, lane = tid & 31;
  const int nqt = T_ / 64;
  int qt = blockIdx.x % nqt;
  int h  = (blockIdx.x / nqt) % H_;
  int b  = blockIdx.x / (nqt * H_);
  int qbase = qt * 64;
  int len = lengths[b];
  int hi8 = (lane >> 4) << 3;
  int nlo = lane & 15;
  int koff = hi8;

  // Q fragments (16 rows x 64 d) held in registers for the whole block
  const u16* qrowp = qkv + (size_t)(b * T_ + qbase + wave * 16 + nlo) * D3_ + h * DH_;
  v16bf qf[2];
#pragma unroll
  for (int d0 = 0; d0 < 2; ++d0) {
    const u16* p = qrowp + d0 * 32 + koff;
    qf[d0] = make_frag(*(const uint4*)p, *(const uint4*)(p + 16));
  }

  v8f o[4] = {};
  float mrow[8], srow[8];
#pragma unroll
  for (int r = 0; r < 8; ++r) { mrow[r] = -FLT_MAX; srow[r] = 0.f; }
  const float scale = 0.125f;  // 1/sqrt(64)

  int nkt = (qbase + 64) / 32;  // causal: only key tiles up to block's last row
  for (int kt = 0; kt < nkt; ++kt) {
    int key0 = kt * 32;
    __syncthreads();
    // stage V tile transposed: Vt[d][key]
#pragma unroll
    for (int i = 0; i < 2; ++i) {
      int flat = (tid + 128 * i) * 8;
      int key = flat >> 6, d = flat & 63;
      const u16* vp = qkv + (size_t)(b * T_ + key0 + key) * D3_ + 2 * D_ + h * DH_ + d;
      uint4 dd = *(const uint4*)vp;
      u32 w4[4] = { dd.x, dd.y, dd.z, dd.w };
#pragma unroll
      for (int jj = 0; jj < 4; ++jj) {
        Vt[d + 2 * jj + 0][key] = (u16)w4[jj];
        Vt[d + 2 * jj + 1][key] = (u16)(w4[jj] >> 16);
      }
    }
    __syncthreads();

    // S = Q @ K^T  (two 16-key subtiles, two 32-deep k-steps over DH=64)
    v8f S[2];
#pragma unroll
    for (int sub = 0; sub < 2; ++sub) {
      v8f c = {};
#pragma unroll
      for (int d0 = 0; d0 < 2; ++d0) {
        const u16* kp = qkv + (size_t)(b * T_ + key0 + sub * 16 + nlo) * D3_ +
                        D_ + h * DH_ + d0 * 32 + koff;
        v16bf kf = make_frag(*(const uint4*)kp, *(const uint4*)(kp + 16));
        c = wmma_bf16(qf[d0], kf, c);
      }
      S[sub] = c;
    }

    // wave-uniform fast path: tile fully unmasked (below diagonal, in-length)
    bool interior = (key0 + 31 <= qbase + wave * 16) && (key0 + 31 < len);

    // online softmax (rows striped: VGPR r -> row r+hi8; lane&15 -> key col)
    float p[2][8];
#pragma unroll
    for (int r = 0; r < 8; ++r) {
      float sv[2];
      if (interior) {
        sv[0] = S[0][r] * scale;
        sv[1] = S[1][r] * scale;
      } else {
        int row = qbase + wave * 16 + r + hi8;
#pragma unroll
        for (int sub = 0; sub < 2; ++sub) {
          int key = key0 + sub * 16 + nlo;
          bool ok = (key <= row) && (key < len);
          sv[sub] = ok ? S[sub][r] * scale : -FLT_MAX;
        }
      }
      float tm = fmaxf(sv[0], sv[1]);
#pragma unroll
      for (int mk = 1; mk < 16; mk <<= 1) tm = fmaxf(tm, __shfl_xor(tm, mk, 16));
      float mnew = fmaxf(mrow[r], tm);
      float corr = __expf(mrow[r] - mnew);
      float ps = 0.f;
#pragma unroll
      for (int sub = 0; sub < 2; ++sub) {
        float pv = (sv[sub] > -FLT_MAX) ? __expf(sv[sub] - mnew) : 0.f;
        p[sub][r] = pv;
        ps += pv;
      }
#pragma unroll
      for (int mk = 1; mk < 16; mk <<= 1) ps += __shfl_xor(ps, mk, 16);
      srow[r] = srow[r] * corr + ps;
      mrow[r] = mnew;
#pragma unroll
      for (int dt = 0; dt < 4; ++dt) o[dt][r] *= corr;
    }

    // relayout P (C-layout f32 -> A-layout bf16) via per-wave LDS scratch
#pragma unroll
    for (int sub = 0; sub < 2; ++sub)
#pragma unroll
      for (int r = 0; r < 8; ++r)
        Psc[wave][r + hi8][sub * 16 + nlo] = f2bf(p[sub][r]);
    asm volatile("s_wait_dscnt 0x0" ::: "memory");  // wave-internal DS RAW
    v16bf pf = load_frag(&Psc[wave][0][0], 40);

    // O += P @ V
#pragma unroll
    for (int dt = 0; dt < 4; ++dt) {
      v16bf vf = load_frag(&Vt[dt * 16][0], 40);
      o[dt] = wmma_bf16(pf, vf, o[dt]);
    }
  }

  // normalize + store bf16 attention output
#pragma unroll
  for (int r = 0; r < 8; ++r) {
    float inv = 1.f / srow[r];
    int row = qbase + wave * 16 + r + hi8;
#pragma unroll
    for (int dt = 0; dt < 4; ++dt)
      obf[(size_t)(b * T_ + row) * D_ + h * DH_ + dt * 16 + nlo] = f2bf(o[dt][r] * inv);
  }
}

// ---------------- orchestration ----------------

extern "C" void kernel_launch(void* const* d_in, const int* in_sizes, int n_in,
                              void* d_out, int out_size, void* d_ws, size_t ws_size,
                              hipStream_t stream) {
  (void)in_sizes; (void)n_in; (void)out_size; (void)ws_size;
  const float* x_in  = (const float*)d_in[0];
  const int*   lens  = (const int*)d_in[1];
  const float* ln0_g = (const float*)d_in[2];
  const float* ln0_b = (const float*)d_in[3];
  const float* Wqkv  = (const float*)d_in[4];
  const float* bqkv  = (const float*)d_in[5];
  const float* Wo    = (const float*)d_in[6];
  const float* bo    = (const float*)d_in[7];
  const float* ln1_g = (const float*)d_in[8];
  const float* ln1_b = (const float*)d_in[9];
  const float* ln2_g = (const float*)d_in[10];
  const float* ln2_b = (const float*)d_in[11];
  const float* W1    = (const float*)d_in[12];
  const float* b1    = (const float*)d_in[13];
  const float* W2    = (const float*)d_in[14];
  const float* b2    = (const float*)d_in[15];

  char* ws = (char*)d_ws;
  size_t off = 0;
  float* x     = (float*)(ws + off); off += (size_t)M_ * D_ * 4;    // 16 MB
  u16* h_bf    = (u16*)(ws + off);   off += (size_t)M_ * D_ * 2;    // 8 MB
  u16* qkv_bf  = (u16*)(ws + off);   off += (size_t)M_ * D3_ * 2;   // 24 MB
  u16* o_bf    = (u16*)(ws + off);   off += (size_t)M_ * D_ * 2;    // 8 MB
  u16* ff_bf   = (u16*)(ws + off);   off += (size_t)M_ * FF_ * 2;   // 32 MB
  u16* wt_buf  = (u16*)(ws + off);   off += (size_t)D_ * FF_ * 2;   // 8 MB (max N*K)

  ln0_pe_kernel<<<M_, 256, 0, stream>>>(x_in, ln0_g, ln0_b, x);

  for (int i = 0; i < L_; ++i) {
    // h = LN1(x) -> bf16
    ln_bf_kernel<<<M_, 256, 0, stream>>>(x, ln1_g + i * D_, ln1_b + i * D_, h_bf);
    // qkv = h @ Wqkv + bqkv -> bf16
    wt_kernel<<<dim3(D3_ / 64, D_ / 32), 256, 0, stream>>>(
        Wqkv + (size_t)i * D_ * D3_, wt_buf, D_, D3_);
    gemm_bf16_kernel<false, false><<<dim3(D3_ / 128, M_ / 128), 256, 0, stream>>>(
        h_bf, wt_buf, bqkv + (size_t)i * D3_, nullptr, nullptr, qkv_bf, D3_, D_);
    // flash attention -> o bf16
    attn_kernel<<<B_ * H_ * (T_ / 64), 128, 0, stream>>>(qkv_bf, lens, o_bf);
    // x = x + o @ Wo + bo
    wt_kernel<<<dim3(D_ / 64, D_ / 32), 256, 0, stream>>>(
        Wo + (size_t)i * D_ * D_, wt_buf, D_, D_);
    gemm_bf16_kernel<false, true><<<dim3(D_ / 128, M_ / 128), 256, 0, stream>>>(
        o_bf, wt_buf, bo + (size_t)i * D_, x, x, nullptr, D_, D_);
    // h2 = LN2(x) -> bf16
    ln_bf_kernel<<<M_, 256, 0, stream>>>(x, ln2_g + i * D_, ln2_b + i * D_, h_bf);
    // ff = gelu(h2 @ W1 + b1) -> bf16
    wt_kernel<<<dim3(FF_ / 64, D_ / 32), 256, 0, stream>>>(
        W1 + (size_t)i * D_ * FF_, wt_buf, D_, FF_);
    gemm_bf16_kernel<true, false><<<dim3(FF_ / 128, M_ / 128), 256, 0, stream>>>(
        h_bf, wt_buf, b1 + (size_t)i * FF_, nullptr, nullptr, ff_bf, FF_, D_);
    // x = x + ff @ W2 + b2   (last layer writes straight to d_out)
    wt_kernel<<<dim3(D_ / 64, FF_ / 32), 256, 0, stream>>>(
        W2 + (size_t)i * FF_ * D_, wt_buf, FF_, D_);
    float* outp = (i == L_ - 1) ? (float*)d_out : x;
    gemm_bf16_kernel<false, true><<<dim3(D_ / 128, M_ / 128), 256, 0, stream>>>(
        ff_bf, wt_buf, b2 + (size_t)i * D_, x, outp, nullptr, D_, FF_);
  }
}
